// Encoder_41841571398364
// MI455X (gfx1250) — matrix operand order repaired and checked by reading, not compiled, not run on hardware
//
#include <hip/hip_runtime.h>
#include <hip/hip_bf16.h>

// ---------------------------------------------------------------------------
// Transformer encoder (S=2048, E=1024, H=16, D=64, L=6, FF=4096) for gfx1250.
// GEMM-shaped math -> v_wmma_f32_16x16x32_bf16 (bf16 A/B, f32 accum).
// LDS staging of bf16 tiles -> global_load_async_to_lds_b128 (ASYNCcnt),
// double-buffered against WMMA compute. Attention is flash-style.
// ---------------------------------------------------------------------------

#define SEQ 2048
#define EMB 1024
#define NH  16
#define HD  64
#define NL  6
#define FFD 4096

typedef __attribute__((ext_vector_type(16))) __bf16 bf16x16;
typedef __attribute__((ext_vector_type(8)))  float  f32x8;

union FragAB {
    bf16x16 bf;
    uint4 q[2];
    unsigned short u[16];
};

__device__ __forceinline__ unsigned short f2bf(float f) {
    union { float f; unsigned int u; } x; x.f = f;
    unsigned int u = x.u;
    unsigned int r = u + 0x7FFFu + ((u >> 16) & 1u);  // round-to-nearest-even
    return (unsigned short)(r >> 16);
}

__device__ __forceinline__ f32x8 zero8() {
    f32x8 v;
#pragma unroll
    for (int i = 0; i < 8; ++i) v[i] = 0.0f;
    return v;
}

// --- CDNA5 async global->LDS copy (ASYNCcnt path), with sync fallback -------
#if __has_builtin(__builtin_amdgcn_global_load_async_to_lds_b128)
#define HAVE_ASYNC_LDS 1
#else
#define HAVE_ASYNC_LDS 0
#endif

// Builtin expects pointers to 16-byte int vectors in AS1 (global) / AS3 (LDS).
typedef int v4i __attribute__((vector_size(16)));
typedef __attribute__((address_space(1))) v4i* gv4i_p;
typedef __attribute__((address_space(3))) v4i* lv4i_p;

__device__ __forceinline__ void async_copy_b128(void* lds_dst, const void* gsrc) {
#if HAVE_ASYNC_LDS
    __builtin_amdgcn_global_load_async_to_lds_b128(
        (gv4i_p)(unsigned long long)(uintptr_t)gsrc,
        (lv4i_p)(unsigned int)(uintptr_t)lds_dst,
        0, 0);
#else
    *(uint4*)lds_dst = *(const uint4*)gsrc;
#endif
}

__device__ __forceinline__ void async_wait0() {
    // Valid gfx1250 SOPP regardless of which copy path was compiled in.
    asm volatile("s_wait_asynccnt 0x0" ::: "memory");
}

// ---------------------------------------------------------------------------
// Tiled WMMA GEMM:  C[M,N] = act(A[M,K](bf16) * B[K,N](f32 weights) + bias)
//   BT=true : B stored as [N,K] row-major (shared projection weights Wk/Wq/Wv)
//   BT=false: B stored as [K,N] row-major (hW*, Wo, W1, W2)
// Block: 256 threads (8 waves), tile 128(M) x 64(N), BK=32.
// A-tiles are async-DMA'd into double-buffered LDS; B-tiles are converted
// f32->bf16 through registers (with prefetch of the next K-tile).
// blockIdx.z: which = z/zdiv selects one of 3 operand sets (merged launches),
//             zi = z%zdiv is the batch index (per-head) applied via *Z strides.
// ---------------------------------------------------------------------------
template<bool BT, bool RELU, bool OUT_BF16>
__global__ void __launch_bounds__(256) gemm_ws(
    const unsigned short* __restrict__ A0, const unsigned short* __restrict__ A1,
    const unsigned short* __restrict__ A2,
    const float* __restrict__ B0, const float* __restrict__ B1,
    const float* __restrict__ B2,
    const float* __restrict__ bias0, const float* __restrict__ bias1,
    const float* __restrict__ bias2,
    void* __restrict__ C0, void* __restrict__ C1, void* __restrict__ C2,
    int M, int N, int K, int lda, int ldb, int ldc,
    int zdiv, long bZ, long biasZ, long cZ)
{
    __shared__ alignas(16) unsigned short As[2][128 * 32];
    __shared__ alignas(16) unsigned short Bs[64 * 32];

    const int t     = threadIdx.x;
    const int zb    = blockIdx.z;
    const int which = zb / zdiv;
    const int zi    = zb % zdiv;

    const unsigned short* A = which == 0 ? A0 : (which == 1 ? A1 : A2);
    const float* B = (which == 0 ? B0 : (which == 1 ? B1 : B2)) + (size_t)zi * bZ;
    const float* bias = which == 0 ? bias0 : (which == 1 ? bias1 : bias2);
    if (bias) bias += (size_t)zi * biasZ;
    void* C = which == 0 ? C0 : (which == 1 ? C1 : C2);

    const int m0   = blockIdx.x * 128;
    const int n0   = blockIdx.y * 64;
    const int lane = t & 31;
    const int wave = t >> 5;
    const int hi   = (lane >> 4) & 1;
    const int lm   = lane & 15;

    f32x8 acc[4];
#pragma unroll
    for (int j = 0; j < 4; ++j) acc[j] = zero8();

    const int arow = t >> 1;
    const int akc  = (t & 1) * 16;
    const int nKT  = K >> 5;

    auto issueA = [&](int kt, int buf) {
        const unsigned short* src = A + (size_t)(m0 + arow) * lda + kt * 32 + akc;
        unsigned short* dst = &As[buf][arow * 32 + akc];
        async_copy_b128(dst, src);
        async_copy_b128(dst + 8, src + 8);
    };

    issueA(0, 0);
    int buf = 0;

    for (int kt = 0; kt < nKT; ++kt) {
        // --- stage B as Bs[n][k], converting f32 -> bf16 ---
        if (BT) {
            const int bn = t >> 2;
            const int bk = (t & 3) * 8;
            const float* sp = B + (size_t)(n0 + bn) * ldb + kt * 32 + bk;
            const float4* src = (const float4*)sp;
            float4 f0 = src[0], f1 = src[1];
            if (kt + 1 < nKT) __builtin_prefetch(sp + 32, 0, 0);
            unsigned short* dst = &Bs[bn * 32 + bk];
            dst[0] = f2bf(f0.x); dst[1] = f2bf(f0.y); dst[2] = f2bf(f0.z); dst[3] = f2bf(f0.w);
            dst[4] = f2bf(f1.x); dst[5] = f2bf(f1.y); dst[6] = f2bf(f1.z); dst[7] = f2bf(f1.w);
        } else {
            const int bk = t >> 3;
            const int bn = (t & 7) * 8;
            const float* sp = B + (size_t)(kt * 32 + bk) * ldb + n0 + bn;
            const float4* src = (const float4*)sp;
            float4 f0 = src[0], f1 = src[1];
            if (kt + 1 < nKT) __builtin_prefetch(sp + (size_t)32 * ldb, 0, 0);
            float f[8] = {f0.x, f0.y, f0.z, f0.w, f1.x, f1.y, f1.z, f1.w};
#pragma unroll
            for (int j = 0; j < 8; ++j) Bs[(bn + j) * 32 + bk] = f2bf(f[j]);
        }

        async_wait0();        // A(kt) landed in LDS
        __syncthreads();      // B stores + everyone's async A visible
        if (kt + 1 < nKT) issueA(kt + 1, buf ^ 1);  // overlap next A with WMMA

        // --- A fragment (ISA 16-bit A 16x32 layout) ---
        FragAB a;
        {
            const int r = (wave * 16 + lm) * 32;
            a.q[0] = *(const uint4*)&As[buf][r + hi * 8];
            a.q[1] = *(const uint4*)&As[buf][r + 16 + hi * 8];
        }
#pragma unroll
        for (int j = 0; j < 4; ++j) {
            FragAB b;
            const uint4* p = (const uint4*)&Bs[(16 * j + lm) * 32 + 8 * hi];
            b.q[0] = p[0]; b.q[1] = p[2];
            acc[j] = __builtin_amdgcn_wmma_f32_16x16x32_bf16(
                false, a.bf, false, b.bf, (short)0, acc[j], false, false);
        }
        __syncthreads();      // done reading Bs before it is restaged
        buf ^= 1;
    }

    // --- epilogue: bias, activation, store ---
#pragma unroll
    for (int j = 0; j < 4; ++j) {
        const int col = n0 + 16 * j + lm;
        const float bv = bias ? bias[col] : 0.0f;
#pragma unroll
        for (int i = 0; i < 8; ++i) {
            const int row = m0 + wave * 16 + i + 8 * hi;  // C layout: M = i + 8*hi
            float v = acc[j][i] + bv;
            if (RELU) v = v > 0.0f ? v : 0.0f;
            if (OUT_BF16)
                ((unsigned short*)C)[(size_t)zi * cZ + (size_t)row * ldc + col] = f2bf(v);
            else
                ((float*)C)[(size_t)zi * cZ + (size_t)row * ldc + col] = v;
        }
    }
}

// ---------------------------------------------------------------------------
// Flash attention. grid = (SEQ/64, NH), block = 128 (4 waves).
// Each wave owns 16 query rows; streams over 64-key tiles with online softmax.
// K tiles are async-DMA'd into LDS; V is transposed through registers.
// qh/kh/vh: [NH, SEQ, HD] bf16. cat: [SEQ, EMB] bf16 (head-concat output).
// ---------------------------------------------------------------------------
__global__ void __launch_bounds__(128) flash_attn(
    const unsigned short* __restrict__ qh, const unsigned short* __restrict__ kh,
    const unsigned short* __restrict__ vh, unsigned short* __restrict__ cat)
{
    __shared__ alignas(16) unsigned short Ks[64 * 64];      // [key][d]
    __shared__ alignas(16) unsigned short Vt[64 * 64];      // [d][key]
    __shared__ alignas(16) unsigned short Ps[4][16 * 64];   // per-wave P scratch

    const int t    = threadIdx.x;
    const int h    = blockIdx.y;
    const int q0   = blockIdx.x * 64;
    const int lane = t & 31;
    const int wave = t >> 5;
    const int hi   = (lane >> 4) & 1;
    const int lm   = lane & 15;
    const float scale = 0.125f;  // 1/sqrt(64)

    // Q fragments: A-layout, read directly from global (16B chunks)
    FragAB qf[2];
    {
        const unsigned short* qrow = qh + ((size_t)h * SEQ + (q0 + wave * 16 + lm)) * HD;
#pragma unroll
        for (int c = 0; c < 2; ++c) {
            qf[c].q[0] = *(const uint4*)(qrow + c * 32 + hi * 8);
            qf[c].q[1] = *(const uint4*)(qrow + c * 32 + 16 + hi * 8);
        }
    }

    float mI[8], lI[8];
    f32x8 o[4];
#pragma unroll
    for (int i = 0; i < 8; ++i) { mI[i] = -3.0e38f; lI[i] = 0.0f; }
#pragma unroll
    for (int j = 0; j < 4; ++j) o[j] = zero8();

    for (int kt = 0; kt < SEQ / 64; ++kt) {
        const int k0 = kt * 64;
        // stage K tile [key][d] via async DMA to LDS
        {
            const int row = t >> 1;
            const int c   = (t & 1) * 32;
            const unsigned short* src = kh + ((size_t)h * SEQ + k0 + row) * HD + c;
            unsigned short* dst = &Ks[row * 64 + c];
            async_copy_b128(dst,      src);
            async_copy_b128(dst + 8,  src + 8);
            async_copy_b128(dst + 16, src + 16);
            async_copy_b128(dst + 24, src + 24);
        }
        // stage V transposed [d][key] (through registers)
        {
            const int key = t & 63;
            const int d0  = (t >> 6) * 32;
            const unsigned short* src = vh + ((size_t)h * SEQ + k0 + key) * HD + d0;
            uint4 r[4];
            r[0] = ((const uint4*)src)[0]; r[1] = ((const uint4*)src)[1];
            r[2] = ((const uint4*)src)[2]; r[3] = ((const uint4*)src)[3];
            const unsigned short* rs = (const unsigned short*)r;
#pragma unroll
            for (int j = 0; j < 32; ++j) Vt[(d0 + j) * 64 + key] = rs[j];
        }
        async_wait0();
        __syncthreads();

        // S = scale * Q K^T  (B[k=d][n=key] = Ks[key][d] -> already [n][k])
        f32x8 sf[4];
#pragma unroll
        for (int j = 0; j < 4; ++j) sf[j] = zero8();
#pragma unroll
        for (int j = 0; j < 4; ++j)
#pragma unroll
            for (int c = 0; c < 2; ++c) {
                FragAB b;
                const uint4* p = (const uint4*)&Ks[(16 * j + lm) * 64 + c * 32 + 8 * hi];
                b.q[0] = p[0]; b.q[1] = p[2];
                sf[j] = __builtin_amdgcn_wmma_f32_16x16x32_bf16(
                    false, qf[c].bf, false, b.bf, (short)0, sf[j], false, false);
            }
#pragma unroll
        for (int j = 0; j < 4; ++j)
#pragma unroll
            for (int i = 0; i < 8; ++i) sf[j][i] *= scale;

        // row max over the 64-key tile (lanes within a 16-lane half hold row's cols)
        float mt[8];
#pragma unroll
        for (int i = 0; i < 8; ++i)
            mt[i] = fmaxf(fmaxf(sf[0][i], sf[1][i]), fmaxf(sf[2][i], sf[3][i]));
#pragma unroll
        for (int off = 1; off < 16; off <<= 1)
#pragma unroll
            for (int i = 0; i < 8; ++i) mt[i] = fmaxf(mt[i], __shfl_xor(mt[i], off, 32));

        float al[8], rs[8];
#pragma unroll
        for (int i = 0; i < 8; ++i) {
            const float mn = fmaxf(mI[i], mt[i]);
            al[i] = __expf(mI[i] - mn);
            mI[i] = mn;
            rs[i] = 0.0f;
        }
#pragma unroll
        for (int j = 0; j < 4; ++j)
#pragma unroll
            for (int i = 0; i < 8; ++i) {
                const float p = __expf(sf[j][i] - mI[i]);
                sf[j][i] = p;
                rs[i] += p;
            }
#pragma unroll
        for (int off = 1; off < 16; off <<= 1)
#pragma unroll
            for (int i = 0; i < 8; ++i) rs[i] += __shfl_xor(rs[i], off, 32);
#pragma unroll
        for (int i = 0; i < 8; ++i) lI[i] = lI[i] * al[i] + rs[i];
#pragma unroll
        for (int j = 0; j < 4; ++j)
#pragma unroll
            for (int i = 0; i < 8; ++i) o[j][i] *= al[i];

        // P: C-layout -> per-wave LDS -> A-layout fragments
#pragma unroll
        for (int j = 0; j < 4; ++j)
#pragma unroll
            for (int i = 0; i < 8; ++i)
                Ps[wave][(i + 8 * hi) * 64 + 16 * j + lm] = f2bf(sf[j][i]);

        FragAB pf[2];
#pragma unroll
        for (int c = 0; c < 2; ++c) {
            pf[c].q[0] = *(const uint4*)&Ps[wave][lm * 64 + c * 32 + hi * 8];
            pf[c].q[1] = *(const uint4*)&Ps[wave][lm * 64 + c * 32 + 16 + hi * 8];
        }

        // O += P @ V  (B[k=key][n=d] = Vt[d][key] -> [n][k])
#pragma unroll
        for (int j = 0; j < 4; ++j)
#pragma unroll
            for (int c = 0; c < 2; ++c) {
                FragAB b;
                const uint4* p = (const uint4*)&Vt[(16 * j + lm) * 64 + c * 32 + 8 * hi];
                b.q[0] = p[0]; b.q[1] = p[2];
                o[j] = __builtin_amdgcn_wmma_f32_16x16x32_bf16(
                    false, pf[c].bf, false, b.bf, (short)0, o[j], false, false);
            }
        __syncthreads();
    }

    // normalize and write concat output (bf16)
#pragma unroll
    for (int i = 0; i < 8; ++i) lI[i] = 1.0f / lI[i];
#pragma unroll
    for (int j = 0; j < 4; ++j)
#pragma unroll
        for (int i = 0; i < 8; ++i) {
            const int srow = q0 + wave * 16 + i + 8 * hi;
            cat[(size_t)srow * EMB + h * HD + 16 * j + lm] = f2bf(o[j][i] * lI[i]);
        }
}

// ---------------------------------------------------------------------------
// Residual + LayerNorm (one row per block). Writes fp32 and bf16 copies.
// ---------------------------------------------------------------------------
__global__ void __launch_bounds__(256) ln_residual(
    const float* __restrict__ src, const float* __restrict__ res,
    const float* __restrict__ g, const float* __restrict__ b,
    float* __restrict__ outf, unsigned short* __restrict__ outb)
{
    __shared__ float red[256];
    const int row = blockIdx.x, t = threadIdx.x;
    const size_t base = (size_t)row * EMB;
    float v[4];
    float sum = 0.0f;
#pragma unroll
    for (int i = 0; i < 4; ++i) {
        const int idx = t + 256 * i;
        v[i] = src[base + idx] + res[base + idx];
        sum += v[i];
    }
    red[t] = sum; __syncthreads();
    for (int s2 = 128; s2 > 0; s2 >>= 1) { if (t < s2) red[t] += red[t + s2]; __syncthreads(); }
    const float mu = red[0] * (1.0f / EMB);
    __syncthreads();
    float sq = 0.0f;
#pragma unroll
    for (int i = 0; i < 4; ++i) { const float d = v[i] - mu; sq += d * d; }
    red[t] = sq; __syncthreads();
    for (int s2 = 128; s2 > 0; s2 >>= 1) { if (t < s2) red[t] += red[t + s2]; __syncthreads(); }
    const float rstd = rsqrtf(red[0] * (1.0f / EMB) + 1e-5f);
#pragma unroll
    for (int i = 0; i < 4; ++i) {
        const int idx = t + 256 * i;
        const float o = (v[i] - mu) * rstd * g[idx] + b[idx];
        outf[base + idx] = o;
        outb[base + idx] = f2bf(o);
    }
}

// ---------------------------------------------------------------------------
// h = x + positional embedding (even rows -> sin, odd rows -> cos)
// ---------------------------------------------------------------------------
__global__ void __launch_bounds__(256) pos_add(
    const float* __restrict__ x, float* __restrict__ hf, unsigned short* __restrict__ hb)
{
    const int s = blockIdx.x, t = threadIdx.x;
    const size_t base = (size_t)s * EMB;
    const bool even = (s & 1) == 0;
    const float ln1e4_over_E = 9.210340371976184f / (float)EMB;
#pragma unroll
    for (int i = 0; i < 4; ++i) {
        const int e = t + 256 * i;
        const float ang = (float)s * __expf(-(float)e * ln1e4_over_E);
        const float pe  = even ? __sinf(ang) : __cosf(ang);
        const float v   = x[base + e] + pe;
        hf[base + e] = v;
        hb[base + e] = f2bf(v);
    }
}

// ---------------------------------------------------------------------------
extern "C" void kernel_launch(void* const* d_in, const int* in_sizes, int n_in,
                              void* d_out, int out_size, void* d_ws, size_t ws_size,
                              hipStream_t stream)
{
    (void)in_sizes; (void)n_in; (void)out_size; (void)ws_size;

    const float* x    = (const float*)d_in[0];
    const float* Wk   = (const float*)d_in[1];
    const float* Wq   = (const float*)d_in[2];
    const float* Wv   = (const float*)d_in[3];
    const float* hWk  = (const float*)d_in[4];
    const float* hbk  = (const float*)d_in[5];
    const float* hWq  = (const float*)d_in[6];
    const float* hbq  = (const float*)d_in[7];
    const float* hWv  = (const float*)d_in[8];
    const float* hbv  = (const float*)d_in[9];
    const float* Wo   = (const float*)d_in[10];
    const float* bo   = (const float*)d_in[11];
    const float* ln1g = (const float*)d_in[12];
    const float* ln1b = (const float*)d_in[13];
    const float* W1   = (const float*)d_in[14];
    const float* b1   = (const float*)d_in[15];
    const float* W2   = (const float*)d_in[16];
    const float* b2   = (const float*)d_in[17];
    const float* ln2g = (const float*)d_in[18];
    const float* ln2b = (const float*)d_in[19];

    char* ws = (char*)d_ws;
    size_t off = 0;
    auto alloc = [&](size_t bytes) -> void* {
        off = (off + 255) & ~(size_t)255;
        void* p = ws + off;
        off += bytes;
        return p;
    };

    float*          hf   = (float*)alloc((size_t)SEQ * EMB * 4);
    unsigned short* hb   = (unsigned short*)alloc((size_t)SEQ * EMB * 2);
    unsigned short* Kbf  = (unsigned short*)alloc((size_t)SEQ * HD * 2);
    unsigned short* Qbf  = (unsigned short*)alloc((size_t)SEQ * HD * 2);
    unsigned short* Vbf  = (unsigned short*)alloc((size_t)SEQ * HD * 2);
    unsigned short* khb  = (unsigned short*)alloc((size_t)NH * SEQ * HD * 2);
    unsigned short* qhb  = (unsigned short*)alloc((size_t)NH * SEQ * HD * 2);
    unsigned short* vhb  = (unsigned short*)alloc((size_t)NH * SEQ * HD * 2);
    unsigned short* catb = (unsigned short*)alloc((size_t)SEQ * EMB * 2);
    float*          of32 = (float*)alloc((size_t)SEQ * EMB * 4);
    unsigned short* f1b  = (unsigned short*)alloc((size_t)SEQ * FFD * 2);

    // h = x + PE
    pos_add<<<SEQ, 256, 0, stream>>>(x, hf, hb);

    for (int l = 0; l < NL; ++l) {
        // shared projections, merged K|Q|V launch: [S,E] @ W^T -> [S,64] each
        gemm_ws<true, false, true><<<dim3(SEQ/128, 1, 3), 256, 0, stream>>>(
            hb, hb, hb, Wk, Wq, Wv, nullptr, nullptr, nullptr,
            Kbf, Qbf, Vbf,
            SEQ, HD, EMB, EMB, EMB, HD, /*zdiv=*/1, 0, 0, 0);

        // per-head linears, merged K|Q|V x 16 heads: [S,64] @ [64,64] + bias
        gemm_ws<false, false, true><<<dim3(SEQ/128, 1, 48), 256, 0, stream>>>(
            Kbf, Qbf, Vbf,
            hWk + (size_t)l * NH * HD * HD, hWq + (size_t)l * NH * HD * HD,
            hWv + (size_t)l * NH * HD * HD,
            hbk + (size_t)l * NH * HD, hbq + (size_t)l * NH * HD,
            hbv + (size_t)l * NH * HD,
            khb, qhb, vhb,
            SEQ, HD, HD, HD, HD, HD, /*zdiv=*/16,
            (long)HD * HD, HD, (long)SEQ * HD);

        // flash attention -> head-concat [S, E] bf16
        flash_attn<<<dim3(SEQ/64, NH, 1), 128, 0, stream>>>(qhb, khb, vhb, catb);

        // output projection: cat @ Wo[l] + bo[l] -> f32
        {
            const float* WoL = Wo + (size_t)l * EMB * EMB;
            const float* boL = bo + (size_t)l * EMB;
            gemm_ws<false, false, false><<<dim3(SEQ/128, EMB/64, 1), 256, 0, stream>>>(
                catb, catb, catb, WoL, WoL, WoL, boL, boL, boL,
                of32, of32, of32,
                SEQ, EMB, EMB, EMB, EMB, EMB, 1, 0, 0, 0);
        }

        // h = LN(o + h)
        ln_residual<<<SEQ, 256, 0, stream>>>(
            of32, hf, ln1g + (size_t)l * EMB, ln1b + (size_t)l * EMB, hf, hb);

        // FFN: relu(h @ W1 + b1) -> bf16
        {
            const float* W1L = W1 + (size_t)l * EMB * FFD;
            const float* b1L = b1 + (size_t)l * FFD;
            gemm_ws<false, true, true><<<dim3(SEQ/128, FFD/64, 1), 256, 0, stream>>>(
                hb, hb, hb, W1L, W1L, W1L, b1L, b1L, b1L,
                f1b, f1b, f1b,
                SEQ, FFD, EMB, EMB, FFD, FFD, 1, 0, 0, 0);
        }

        // f1 @ W2 + b2 -> f32
        {
            const float* W2L = W2 + (size_t)l * FFD * EMB;
            const float* b2L = b2 + (size_t)l * EMB;
            gemm_ws<false, false, false><<<dim3(SEQ/128, EMB/64, 1), 256, 0, stream>>>(
                f1b, f1b, f1b, W2L, W2L, W2L, b2L, b2L, b2L,
                of32, of32, of32,
                SEQ, EMB, FFD, FFD, EMB, EMB, 1, 0, 0, 0);
        }

        // h = LN(f + h); final layer writes fp32 result straight into d_out
        float* lnout = (l == NL - 1) ? (float*)d_out : hf;
        ln_residual<<<SEQ, 256, 0, stream>>>(
            of32, hf, ln2g + (size_t)l * EMB, ln2b + (size_t)l * EMB, lnout, hb);
    }
}
